// E68SelfGatingInverseCell_4887672783144
// MI455X (gfx1250) — matrix operand order repaired and checked by reading, not compiled, not run on hardware
//
#include <hip/hip_runtime.h>

// E68 SelfGatingInverseCell for MI455X (gfx1250, wave32).
//
// Phase 0 (if ws_size >= 8MB): split W_alpha/W_x into bf16 hi/lo planes.
// Phase 1: fused dual GEMM (alpha & v_raw share the A operand):
//   - preferred: bf16x3 split-precision via v_wmma_f32_16x16x32_bf16
//     (3 WMMAs of K=32 replace 8 fp32 WMMAs of K=4 -> ~2.7x matrix-op cycles)
//   - fallback:  exact fp32 via v_wmma_f32_16x16x4_f32
//   Activation epilogue fused; alpha staged in d_out's `outs` region,
//   v_raw in the h[1..T] region.
// Phase 2: per-element recurrence scan (16384 independent recurrences),
//   overwrites the staged values with the final results in place.

typedef __attribute__((ext_vector_type(2)))  float  v2f;
typedef __attribute__((ext_vector_type(4)))  float  v4f;
typedef __attribute__((ext_vector_type(8)))  float  v8f;
typedef __attribute__((ext_vector_type(16))) __bf16 v16bf;

#define Tn   2048
#define Bn   16
#define Dn   1024
#define Mtot (Tn * Bn)          // 32768
#define BD   (Bn * Dn)          // 16384
#define NCHUNK8  (Dn / 8)       // 128 chunks for fp32 path
#define NCHUNK32 (Dn / 32)      // 32 chunks for bf16x3 path

static __device__ __forceinline__ v8f wmma_f32(v2f a, v2f b, v8f c) {
    return __builtin_amdgcn_wmma_f32_16x16x4_f32(
        false, a, false, b, (short)0, c, false, false);
}
static __device__ __forceinline__ v8f wmma_bf16(v16bf a, v16bf b, v8f c) {
    return __builtin_amdgcn_wmma_f32_16x16x32_bf16(
        false, a, false, b, (short)0, c, false, false);
}

// ---------------------------------------------------------------------------
// Phase 0: split both weight matrices into bf16 hi/lo planes (row-major).
// ---------------------------------------------------------------------------
__global__ __launch_bounds__(256) void split_w_kernel(
    const float* __restrict__ Wa, const float* __restrict__ Wx,
    __bf16* __restrict__ wa_hi, __bf16* __restrict__ wa_lo,
    __bf16* __restrict__ wx_hi, __bf16* __restrict__ wx_lo)
{
    const int i = blockIdx.x * blockDim.x + threadIdx.x;  // 0 .. Dn*Dn-1
    const float a = Wa[i];
    const __bf16 ah = (__bf16)a;
    wa_hi[i] = ah;
    wa_lo[i] = (__bf16)(a - (float)ah);
    const float b = Wx[i];
    const __bf16 bh = (__bf16)b;
    wx_hi[i] = bh;
    wx_lo[i] = (__bf16)(b - (float)bh);
}

// ---------------------------------------------------------------------------
// Phase 1 (preferred): bf16x3 dual GEMM.
// Grid: x = N/64 (16), y = M/128 (256). Block: 256 threads = 8 waves.
// Wave w: rows [by*128+16w, +16) x cols [bx*64, +64) for BOTH gemms.
//
// Fragment mapping (identity localK <-> column k0+localK):
//  A (16-bit 16x32): lo lanes hold K0..15, hi lanes K16..31 (same as B)
//    -> two 32B fp32 segments at k0+16*half and k0+16*half+8, cvt to bf16.
//  B (32x16): lo lanes K0..15, hi lanes K16..31 -> one contiguous 32B
//    bf16 load per plane at col k0+16*half.
// ---------------------------------------------------------------------------
__global__ __launch_bounds__(256) void dual_gemm_bf16x3_kernel(
    const float*  __restrict__ x,
    const __bf16* __restrict__ wa_hi, const __bf16* __restrict__ wa_lo,
    const __bf16* __restrict__ wx_hi, const __bf16* __restrict__ wx_lo,
    const float*  __restrict__ b_alpha, const float* __restrict__ b_v,
    float* __restrict__ alpha_out, float* __restrict__ vraw_out)
{
    const int lane = threadIdx.x & 31;
    const int wave = threadIdx.x >> 5;
    const int l16  = lane & 15;
    const int hi   = lane >> 4;

    const int m0 = blockIdx.y * 128 + wave * 16;
    const int n0 = blockIdx.x * 64;

    // A segment bases (fp32): row m0+l16.
    const float* xp0 = x + (long)(m0 + l16) * Dn + 16 * hi;      // seg0
    const float* xp1 = xp0 + 8;                                   // seg1
    // B fragment element offsets (bf16 planes, row-major [Dn,Dn]).
    long boff[4];
#pragma unroll
    for (int s = 0; s < 4; ++s)
        boff[s] = (long)(n0 + 16 * s + l16) * Dn + 16 * hi;

    v8f accA[4], accX[4];
#pragma unroll
    for (int s = 0; s < 4; ++s) { accA[s] = (v8f)(0.0f); accX[s] = (v8f)(0.0f); }

#pragma unroll
    for (int c = 0; c < NCHUNK32; ++c) {
        const int k0 = c * 32;
        // ---- A: load 16 fp32, split to bf16 hi/lo fragments ----
        const v4f a0 = *(const v4f*)(xp0 + k0);
        const v4f a1 = *(const v4f*)(xp0 + k0 + 4);
        const v4f a2 = *(const v4f*)(xp1 + k0);
        const v4f a3 = *(const v4f*)(xp1 + k0 + 4);
        float f[16];
#pragma unroll
        for (int e = 0; e < 4; ++e) {
            f[e]      = a0[e];
            f[e + 4]  = a1[e];
            f[e + 8]  = a2[e];
            f[e + 12] = a3[e];
        }
        v16bf ahi, alo;
#pragma unroll
        for (int e = 0; e < 16; ++e) {
            const __bf16 h = (__bf16)f[e];
            ahi[e] = h;
            alo[e] = (__bf16)(f[e] - (float)h);
        }
        // ---- B fragments + 3 WMMAs per (subtile, matrix) ----
#pragma unroll
        for (int s = 0; s < 4; ++s) {
            const v16bf bAh = *(const v16bf*)(wa_hi + boff[s] + k0);
            const v16bf bAl = *(const v16bf*)(wa_lo + boff[s] + k0);
            accA[s] = wmma_bf16(ahi, bAh, accA[s]);
            accA[s] = wmma_bf16(alo, bAh, accA[s]);
            accA[s] = wmma_bf16(ahi, bAl, accA[s]);
            const v16bf bXh = *(const v16bf*)(wx_hi + boff[s] + k0);
            const v16bf bXl = *(const v16bf*)(wx_lo + boff[s] + k0);
            accX[s] = wmma_bf16(ahi, bXh, accX[s]);
            accX[s] = wmma_bf16(alo, bXh, accX[s]);
            accX[s] = wmma_bf16(ahi, bXl, accX[s]);
        }
    }

    // Activation epilogue: C/D layout -> (M = r + 8*hi, N = l16) in vgpr r.
#pragma unroll
    for (int s = 0; s < 4; ++s) {
        const int n  = n0 + 16 * s + l16;
        const float ba = b_alpha[n];
        const float bv = b_v[n];
#pragma unroll
        for (int r = 0; r < 8; ++r) {
            const long m = m0 + r + 8 * hi;
            alpha_out[m * Dn + n] = 1.0f / (1.0f + __expf(-(accA[s][r] + ba)));
            vraw_out [m * Dn + n] = tanhf(accX[s][r] + bv);
        }
    }
}

// ---------------------------------------------------------------------------
// Phase 1 (fallback, ws too small): exact fp32 WMMA dual GEMM.
// ---------------------------------------------------------------------------
__global__ __launch_bounds__(256) void dual_gemm_f32_kernel(
    const float* __restrict__ x,
    const float* __restrict__ Wa, const float* __restrict__ Wx,
    const float* __restrict__ b_alpha, const float* __restrict__ b_v,
    float* __restrict__ alpha_out, float* __restrict__ vraw_out)
{
    const int lane = threadIdx.x & 31;
    const int wave = threadIdx.x >> 5;
    const int l16  = lane & 15;
    const int hi   = lane >> 4;

    const int m0 = blockIdx.y * 128 + wave * 16;
    const int n0 = blockIdx.x * 64;

    const float* xp = x + (long)(m0 + l16) * Dn + 4 * hi;
    const float* wap[4];
    const float* wxp[4];
#pragma unroll
    for (int s = 0; s < 4; ++s) {
        const long nrow = (long)(n0 + 16 * s + l16) * Dn + 4 * hi;
        wap[s] = Wa + nrow;
        wxp[s] = Wx + nrow;
    }

    v8f accA[4], accX[4];
#pragma unroll
    for (int s = 0; s < 4; ++s) { accA[s] = (v8f)(0.0f); accX[s] = (v8f)(0.0f); }

    v4f aF[2], bAF[2][4], bXF[2][4];
    aF[0] = *(const v4f*)(xp);
#pragma unroll
    for (int s = 0; s < 4; ++s) {
        bAF[0][s] = *(const v4f*)(wap[s]);
        bXF[0][s] = *(const v4f*)(wxp[s]);
    }

#pragma unroll 2
    for (int c = 0; c < NCHUNK8 - 1; ++c) {
        const int cur = c & 1, nxt = cur ^ 1, kn = (c + 1) * 8;
        aF[nxt] = *(const v4f*)(xp + kn);
#pragma unroll
        for (int s = 0; s < 4; ++s) {
            bAF[nxt][s] = *(const v4f*)(wap[s] + kn);
            bXF[nxt][s] = *(const v4f*)(wxp[s] + kn);
        }
        const v2f a01 = { aF[cur].x, aF[cur].y };
        const v2f a23 = { aF[cur].z, aF[cur].w };
#pragma unroll
        for (int s = 0; s < 4; ++s) {
            accA[s] = wmma_f32(a01, (v2f){bAF[cur][s].x, bAF[cur][s].y}, accA[s]);
            accA[s] = wmma_f32(a23, (v2f){bAF[cur][s].z, bAF[cur][s].w}, accA[s]);
            accX[s] = wmma_f32(a01, (v2f){bXF[cur][s].x, bXF[cur][s].y}, accX[s]);
            accX[s] = wmma_f32(a23, (v2f){bXF[cur][s].z, bXF[cur][s].w}, accX[s]);
        }
    }
    {
        const int cur = (NCHUNK8 - 1) & 1;
        const v2f a01 = { aF[cur].x, aF[cur].y };
        const v2f a23 = { aF[cur].z, aF[cur].w };
#pragma unroll
        for (int s = 0; s < 4; ++s) {
            accA[s] = wmma_f32(a01, (v2f){bAF[cur][s].x, bAF[cur][s].y}, accA[s]);
            accA[s] = wmma_f32(a23, (v2f){bAF[cur][s].z, bAF[cur][s].w}, accA[s]);
            accX[s] = wmma_f32(a01, (v2f){bXF[cur][s].x, bXF[cur][s].y}, accX[s]);
            accX[s] = wmma_f32(a23, (v2f){bXF[cur][s].z, bXF[cur][s].w}, accX[s]);
        }
    }

#pragma unroll
    for (int s = 0; s < 4; ++s) {
        const int n  = n0 + 16 * s + l16;
        const float ba = b_alpha[n];
        const float bv = b_v[n];
#pragma unroll
        for (int r = 0; r < 8; ++r) {
            const long m = m0 + r + 8 * hi;
            alpha_out[m * Dn + n] = 1.0f / (1.0f + __expf(-(accA[s][r] + ba)));
            vraw_out [m * Dn + n] = tanhf(accX[s][r] + bv);
        }
    }
}

// ---------------------------------------------------------------------------
// Phase 2: one thread per (b,d) element; h kept in a register over T steps.
// 64-thread blocks spread the 512 waves across more WGPs.
// ---------------------------------------------------------------------------
__global__ __launch_bounds__(64) void scan_kernel(
    float* __restrict__ outs,   // [T,B,D]; holds alpha on entry
    float* __restrict__ hbuf,   // [T+1,B,D]; h[1..T] holds v_raw on entry
    const float* __restrict__ d_g,
    const float* __restrict__ b_g)
{
    const int i = blockIdx.x * blockDim.x + threadIdx.x;   // 0 .. BD-1
    const int d = i & (Dn - 1);
    const float dg = d_g[d];
    const float bg = b_g[d];

    float h = 0.0f;
    hbuf[i] = 0.0f;   // h[0] = 0

#pragma unroll 4
    for (int t = 0; t < Tn; ++t) {
        const long o0 = (long)t * BD + i;
        const long o1 = (long)(t + 1) * BD + i;
        const float a  = outs[o0];   // alpha
        const float vr = hbuf[o1];   // v_raw
        const float g  = 1.0f / (1.0f + __expf(dg * fabsf(h) - bg));
        const float v  = vr * g;
        h = a * h + (1.0f - a) * v;
        const float sig = 1.0f / (1.0f + __expf(-h));
        outs[o0] = h * h * sig;      // h * silu(h)
        hbuf[o1] = h;
    }
}

extern "C" void kernel_launch(void* const* d_in, const int* in_sizes, int n_in,
                              void* d_out, int out_size, void* d_ws, size_t ws_size,
                              hipStream_t stream) {
    const float* x  = (const float*)d_in[0];
    const float* Wa = (const float*)d_in[1];
    const float* ba = (const float*)d_in[2];
    const float* dg = (const float*)d_in[3];
    const float* bg = (const float*)d_in[4];
    const float* Wx = (const float*)d_in[5];
    const float* bv = (const float*)d_in[6];

    float* out  = (float*)d_out;
    float* outs = out;                          // [T,B,D]
    float* hbuf = out + (long)Tn * Bn * Dn;     // [T+1,B,D]
    float* alpha_stage = outs;                  // alpha staged in outs
    float* vraw_stage  = hbuf + (long)BD;       // v_raw staged in h[1..T]

    const size_t DD   = (size_t)Dn * Dn;
    const size_t need = 4 * DD * sizeof(__bf16);   // 8 MB of bf16 planes

    dim3 grid_gemm(Dn / 64, Mtot / 128);           // (16, 256)

    if (ws_size >= need) {
        __bf16* wsb   = (__bf16*)d_ws;
        __bf16* wa_hi = wsb;
        __bf16* wa_lo = wsb + DD;
        __bf16* wx_hi = wsb + 2 * DD;
        __bf16* wx_lo = wsb + 3 * DD;
        split_w_kernel<<<(int)(DD / 256), 256, 0, stream>>>(
            Wa, Wx, wa_hi, wa_lo, wx_hi, wx_lo);
        dual_gemm_bf16x3_kernel<<<grid_gemm, 256, 0, stream>>>(
            x, wa_hi, wa_lo, wx_hi, wx_lo, ba, bv, alpha_stage, vraw_stage);
    } else {
        dual_gemm_f32_kernel<<<grid_gemm, 256, 0, stream>>>(
            x, Wa, Wx, ba, bv, alpha_stage, vraw_stage);
    }

    scan_kernel<<<BD / 64, 64, 0, stream>>>(outs, hbuf, dg, bg);
}
